// Model_38929583571305
// MI455X (gfx1250) — matrix operand order repaired and checked by reading, not compiled, not run on hardware
//
#include <hip/hip_runtime.h>

typedef __attribute__((ext_vector_type(16))) _Float16 v16h;
typedef __attribute__((ext_vector_type(8)))  _Float16 v8h;
typedef __attribute__((ext_vector_type(8)))  float    v8f;

#define B_      4
#define S_      1024
#define H_      32
#define HK_     8
#define D_      128
#define GROUP_  4
#define TOTAL_  (B_*S_)
#define BM      64
#define BN      64
#define NWAVE   4
#define KPAD    8          // row padding (halves) -> conflict-free b128 LDS reads
#define NEG_INF_F (-1e30f)

// ============================================================================
// Prep kernel 1: K fp32 -> f16, identical layout [token][hk][d]
// ============================================================================
__global__ __launch_bounds__(256) void cvt_k_f16(const float* __restrict__ k,
                                                 _Float16* __restrict__ kh)
{
    const size_t i = ((size_t)blockIdx.x * 256 + threadIdx.x) * 8;
    if (i >= (size_t)TOTAL_ * HK_ * D_) return;
    const float4 a = *(const float4*)(k + i);
    const float4 b = *(const float4*)(k + i + 4);
    v8h o;
    o[0]=(_Float16)a.x; o[1]=(_Float16)a.y; o[2]=(_Float16)a.z; o[3]=(_Float16)a.w;
    o[4]=(_Float16)b.x; o[5]=(_Float16)b.y; o[6]=(_Float16)b.z; o[7]=(_Float16)b.w;
    *(v8h*)(kh + i) = o;
}

// ============================================================================
// Prep kernel 2: V fp32 -> f16 transposed [b][hk][d][s]  (LDS-tiled 64x64)
// ============================================================================
__global__ __launch_bounds__(256) void cvt_v_t_f16(const float* __restrict__ v,
                                                   const int* __restrict__ b_start_loc,
                                                   _Float16* __restrict__ vt)
{
    __shared__ __align__(16) _Float16 sT[64][64];
    const int tid  = threadIdx.x;
    const int tx   = blockIdx.x & 15;   // pos tile (S/64 = 16)
    const int td   = blockIdx.x >> 4;   // d   tile (D/64 = 2)
    const int hk   = blockIdx.y;
    const int b    = blockIdx.z;
    const int pos0 = tx * 64;
    const int d0   = td * 64;
    const int bstart = b_start_loc[b];

#pragma unroll
    for (int it = 0; it < 4; ++it) {
        const int e = (it*256 + tid) * 4;    // 0..4095
        const int p = e >> 6;                // pos within tile
        const int d = e & 63;                // d   within tile
        const float4 f = *(const float4*)(v + (size_t)(bstart + pos0 + p)*(HK_*D_)
                                            + hk*D_ + d0 + d);
        sT[d+0][p]=(_Float16)f.x; sT[d+1][p]=(_Float16)f.y;
        sT[d+2][p]=(_Float16)f.z; sT[d+3][p]=(_Float16)f.w;
    }
    __syncthreads();
    const int row = tid >> 2;                // 0..63 (d within tile)
    const int c   = (tid & 3) * 16;          // 0/16/32/48 (pos within tile)
    _Float16* dst = vt + (((size_t)b*HK_ + hk)*D_ + d0 + row)*S_ + pos0 + c;
    *(v8h*)(dst)     = *(const v8h*)&sT[row][c];
    *(v8h*)(dst + 8) = *(const v8h*)&sT[row][c + 8];
}

// ============================================================================
// Flash-attention forward, causal, GQA(4:1), f16 WMMA, f32 accumulate.
// PRECONV=true: K/V streamed as f16 from workspace (V already transposed).
// ============================================================================
template<bool PRECONV>
__global__ __launch_bounds__(128) void fa_fwd_gfx1250(
    const float* __restrict__ q, const float* __restrict__ k,
    const float* __restrict__ v, const int* __restrict__ b_start_loc,
    const int* __restrict__ b_seq_len, float* __restrict__ out,
    const _Float16* __restrict__ kh, const _Float16* __restrict__ vt)
{
    __shared__ __align__(16) _Float16 sK [BN][D_ + KPAD];       // [key][d]
    __shared__ __align__(16) _Float16 sVT[D_][BN + KPAD];       // [d][key]
    __shared__ __align__(16) _Float16 sP [NWAVE][16][BN + KPAD];

    const int tid  = threadIdx.x;
    const int wv   = tid >> 5;
    const int lane = tid & 31;
    const int half = lane >> 4;
    const int l16  = lane & 15;

    const int m0 = blockIdx.x * BM;
    const int h  = blockIdx.y;
    const int b  = blockIdx.z;
    const int hk = h / GROUP_;

    const int bstart = b_start_loc[b];
    const int seqlen = b_seq_len[b];
    const float sm_scale = 0.08838834764831845f;  // 1/sqrt(128)

    // ---- Q A-fragments (16x32 layout): lane&15 -> M; a[0..7]=K(half*8+j),
    //      a[8..15]=K(16+half*8+j) ----
    const int qrow = m0 + wv*16 + l16;
    const float* qptr = q + (size_t)(bstart + qrow) * (H_*D_) + h*D_;
    v16h qa[4];
#pragma unroll
    for (int kc = 0; kc < 4; ++kc) {
        const int d0 = kc*32 + half*8;
        const float4 f0 = *(const float4*)(qptr + d0);
        const float4 f1 = *(const float4*)(qptr + d0 + 4);
        const float4 f2 = *(const float4*)(qptr + d0 + 16);
        const float4 f3 = *(const float4*)(qptr + d0 + 20);
        v16h a;
        a[0]=(_Float16)f0.x;  a[1]=(_Float16)f0.y;  a[2]=(_Float16)f0.z;  a[3]=(_Float16)f0.w;
        a[4]=(_Float16)f1.x;  a[5]=(_Float16)f1.y;  a[6]=(_Float16)f1.z;  a[7]=(_Float16)f1.w;
        a[8]=(_Float16)f2.x;  a[9]=(_Float16)f2.y;  a[10]=(_Float16)f2.z; a[11]=(_Float16)f2.w;
        a[12]=(_Float16)f3.x; a[13]=(_Float16)f3.y; a[14]=(_Float16)f3.z; a[15]=(_Float16)f3.w;
        qa[kc] = a;
    }

    float m_run[8], l_run[8];
    v8f oacc[8];
#pragma unroll
    for (int r = 0; r < 8; ++r) { m_run[r] = NEG_INF_F; l_run[r] = 0.f; }
#pragma unroll
    for (int ds = 0; ds < 8; ++ds) oacc[ds] = (v8f){};

    const int kv_end = (m0 + BM < seqlen) ? (m0 + BM) : seqlen;

    for (int n0 = 0; n0 < kv_end; n0 += BN) {
        __syncthreads();   // previous tile fully consumed

        if constexpr (PRECONV) {
            // ---- pure f16 tile staging (16 KB K + 16 KB V^T) ----
            const int key = tid >> 1;                 // K: 2 threads per key row
            const int dh  = (tid & 1) * 64;           // halves offset within row
            const _Float16* ksrc = kh + (size_t)(bstart + n0 + key)*(HK_*D_)
                                      + hk*D_ + dh;
            const _Float16* vsrc = vt + (((size_t)b*HK_ + hk)*D_ + tid)*S_ + n0;
#pragma unroll
            for (int i = 0; i < 8; ++i)
                *(v8h*)&sK[key][dh + i*8] = *(const v8h*)(ksrc + i*8);
#pragma unroll
            for (int i = 0; i < 8; ++i)
                *(v8h*)&sVT[tid][i*8] = *(const v8h*)(vsrc + i*8);
            if (n0 + BN < kv_end)   // warm L2 for next tile
                __builtin_prefetch(ksrc + (size_t)BN*(HK_*D_), 0, 1);
        } else {
            // ---- fallback: fp32 load + convert + transpose V in LDS ----
#pragma unroll
            for (int it = 0; it < 16; ++it) {
                const int e   = (it*128 + tid) * 4;
                const int key = e >> 7;
                const int d   = e & (D_ - 1);
                const size_t off = (size_t)(bstart + n0 + key)*(HK_*D_) + hk*D_ + d;
                const float4 kf = *(const float4*)(k + off);
                const float4 vf = *(const float4*)(v + off);
                sK[key][d+0]=(_Float16)kf.x; sK[key][d+1]=(_Float16)kf.y;
                sK[key][d+2]=(_Float16)kf.z; sK[key][d+3]=(_Float16)kf.w;
                sVT[d+0][key]=(_Float16)vf.x; sVT[d+1][key]=(_Float16)vf.y;
                sVT[d+2][key]=(_Float16)vf.z; sVT[d+3][key]=(_Float16)vf.w;
            }
        }
        __syncthreads();

        // ---- S = Q K^T : 4 N-subtiles x 4 K-chunks ----
        v8f sc[4];
#pragma unroll
        for (int n = 0; n < 4; ++n) {
            v8f acc = (v8f){};
            const int key = n*16 + l16;    // B: lane&15 -> N (key)
#pragma unroll
            for (int kc = 0; kc < 4; ++kc) {
                const int dstart = kc*32 + half*8;
                const v8h lo = *(const v8h*)&sK[key][dstart];
                const v8h hi = *(const v8h*)&sK[key][dstart + 16];
                v16h bf;
#pragma unroll
                for (int i = 0; i < 8; ++i) { bf[i] = lo[i]; bf[8+i] = hi[i]; }
                acc = __builtin_amdgcn_wmma_f32_16x16x32_f16(
                          false, qa[kc], false, bf, (short)0, acc, false, false);
            }
            sc[n] = acc;
        }

        // ---- scale + causal/seqlen mask (row = half*8+r, col = n*16+l16) ----
        const bool need_mask = (n0 + BN > m0) || (n0 + BN > seqlen);
#pragma unroll
        for (int n = 0; n < 4; ++n) {
#pragma unroll
            for (int r = 0; r < 8; ++r) {
                float s = sc[n][r] * sm_scale;
                if (need_mask) {
                    const int qp = m0 + wv*16 + half*8 + r;
                    const int kp = n0 + n*16 + l16;
                    if (kp > qp || kp >= seqlen) s = NEG_INF_F;
                }
                sc[n][r] = s;
            }
        }

        // ---- online softmax (row reductions inside 16-lane halves) ----
        float mx[8];
#pragma unroll
        for (int r = 0; r < 8; ++r) {
            float m_ = fmaxf(fmaxf(sc[0][r], sc[1][r]), fmaxf(sc[2][r], sc[3][r]));
#pragma unroll
            for (int s = 1; s < 16; s <<= 1)
                m_ = fmaxf(m_, __shfl_xor(m_, s, 32));
            mx[r] = m_;
        }
        float alpha[8], rsum[8];
#pragma unroll
        for (int r = 0; r < 8; ++r) {
            const float mn = fmaxf(m_run[r], mx[r]);
            alpha[r] = __expf(m_run[r] - mn);
            m_run[r] = mn;
            rsum[r]  = 0.f;
        }
        const int prow = half*8;
#pragma unroll
        for (int n = 0; n < 4; ++n) {
#pragma unroll
            for (int r = 0; r < 8; ++r) {
                const float p = __expf(sc[n][r] - m_run[r]);
                rsum[r] += p;
                sP[wv][prow + r][n*16 + l16] = (_Float16)p;
            }
        }
#pragma unroll
        for (int r = 0; r < 8; ++r) {
#pragma unroll
            for (int s = 1; s < 16; s <<= 1)
                rsum[r] += __shfl_xor(rsum[r], s, 32);
            l_run[r] = l_run[r] * alpha[r] + rsum[r];
        }
#pragma unroll
        for (int ds = 0; ds < 8; ++ds)
#pragma unroll
            for (int r = 0; r < 8; ++r) oacc[ds][r] *= alpha[r];

        // ---- O += P V ----
        v16h pa[2];
#pragma unroll
        for (int c2 = 0; c2 < 2; ++c2) {
            const int kb = c2*32 + half*8;
            const v8h lo = *(const v8h*)&sP[wv][l16][kb];
            const v8h hi = *(const v8h*)&sP[wv][l16][kb + 16];
            v16h a;
#pragma unroll
            for (int i = 0; i < 8; ++i) { a[i] = lo[i]; a[8+i] = hi[i]; }
            pa[c2] = a;
        }
#pragma unroll
        for (int ds = 0; ds < 8; ++ds) {
            const int d = ds*16 + l16;     // B: lane&15 -> N (d column)
#pragma unroll
            for (int c2 = 0; c2 < 2; ++c2) {
                const int kb = c2*32 + half*8;
                const v8h lo = *(const v8h*)&sVT[d][kb];
                const v8h hi = *(const v8h*)&sVT[d][kb + 16];
                v16h bf;
#pragma unroll
                for (int i = 0; i < 8; ++i) { bf[i] = lo[i]; bf[8+i] = hi[i]; }
                oacc[ds] = __builtin_amdgcn_wmma_f32_16x16x32_f16(
                               false, pa[c2], false, bf, (short)0, oacc[ds], false, false);
            }
        }
    }

    // ---- normalize + store ----
    const int orow_base = m0 + wv*16 + half*8;
#pragma unroll
    for (int r = 0; r < 8; ++r) {
        const int qp = orow_base + r;
        const float inv = (qp < seqlen && l_run[r] > 0.f) ? (1.f / l_run[r]) : 0.f;
        float* optr = out + (size_t)(bstart + qp) * (H_*D_) + h*D_;
#pragma unroll
        for (int ds = 0; ds < 8; ++ds)
            optr[ds*16 + l16] = oacc[ds][r] * inv;
    }
}

// ============================================================================
extern "C" void kernel_launch(void* const* d_in, const int* in_sizes, int n_in,
                              void* d_out, int out_size, void* d_ws, size_t ws_size,
                              hipStream_t stream)
{
    const float* q           = (const float*)d_in[0];
    const float* k           = (const float*)d_in[1];
    const float* v           = (const float*)d_in[2];
    const int*   b_start_loc = (const int*)d_in[3];
    const int*   b_seq_len   = (const int*)d_in[4];
    float* out = (float*)d_out;

    const size_t kh_elems = (size_t)TOTAL_ * HK_ * D_;          // 4.19M halves
    const size_t vt_elems = (size_t)B_ * HK_ * D_ * S_;         // 4.19M halves
    const size_t need     = (kh_elems + vt_elems) * sizeof(_Float16);

    dim3 grid(S_ / BM, H_, B_);
    dim3 block(NWAVE * 32);

    if (ws_size >= need) {
        _Float16* kh = (_Float16*)d_ws;
        _Float16* vt = kh + kh_elems;
        cvt_k_f16<<<dim3((unsigned)(kh_elems / 8 / 256)), 256, 0, stream>>>(k, kh);
        cvt_v_t_f16<<<dim3(32, HK_, B_), 256, 0, stream>>>(v, b_start_loc, vt);
        fa_fwd_gfx1250<true><<<grid, block, 0, stream>>>(
            q, k, v, b_start_loc, b_seq_len, out, kh, vt);
    } else {
        fa_fwd_gfx1250<false><<<grid, block, 0, stream>>>(
            q, k, v, b_start_loc, b_seq_len, out, nullptr, nullptr);
    }
}
